// Attention_13941463843545
// MI455X (gfx1250) — compile-verified, matrix operand order checked by reading, not run on hardware
//
#include <hip/hip_runtime.h>

#define DIMC 384
#define NPIX 3136   // 56*56
#define NBATCH 8

typedef __attribute__((ext_vector_type(16))) _Float16     v16h;
typedef __attribute__((ext_vector_type(8)))  float        v8f;
typedef __attribute__((ext_vector_type(4)))  unsigned int v4u;
typedef __attribute__((ext_vector_type(8)))  unsigned int v8u;

static __device__ __forceinline__ v8f wmma16(v16h a, v16h b, v8f c) {
  // D = A(16x32 f16) x B(32x16 f16) + C(16x16 f32)
  return __builtin_amdgcn_wmma_f32_16x16x32_f16(false, a, false, b, (short)0, c,
                                                false, false);
}

// Load one 16x32 f16 fragment (A or B operand) given a pointer already offset
// by (row*stride + kchunk + hb*8) in halfword units.  lo dwords = K{0..7|8..15},
// hi dwords (+16 halves) = K{16..23|24..31} per the CDNA5 16-bit VGPR layout.
static __device__ __forceinline__ v16h load_frag(const _Float16* p) {
  v4u lo = *(const v4u*)(p);
  v4u hi = *(const v4u*)(p + 16);
  v8u u;
  u[0] = lo[0]; u[1] = lo[1]; u[2] = lo[2]; u[3] = lo[3];
  u[4] = hi[0]; u[5] = hi[1]; u[6] = hi[2]; u[7] = hi[3];
  return __builtin_bit_cast(v16h, u);
}

// B fragment for fp32 weight row (cvt f32->f16 on the fly).
static __device__ __forceinline__ v16h wfrag(const float* wr, int hb) {
  const float4* p0 = (const float4*)(wr + hb * 8);
  const float4* p1 = (const float4*)(wr + 16 + hb * 8);
  float4 a0 = p0[0], a1 = p0[1], b0 = p1[0], b1 = p1[1];
  v16h h;
  h[0]  = (_Float16)a0.x; h[1]  = (_Float16)a0.y; h[2]  = (_Float16)a0.z; h[3]  = (_Float16)a0.w;
  h[4]  = (_Float16)a1.x; h[5]  = (_Float16)a1.y; h[6]  = (_Float16)a1.z; h[7]  = (_Float16)a1.w;
  h[8]  = (_Float16)b0.x; h[9]  = (_Float16)b0.y; h[10] = (_Float16)b0.z; h[11] = (_Float16)b0.w;
  h[12] = (_Float16)b1.x; h[13] = (_Float16)b1.y; h[14] = (_Float16)b1.z; h[15] = (_Float16)b1.w;
  return h;
}

static __device__ __forceinline__ unsigned int pack2(float a, float b) {
  unsigned short ua = __builtin_bit_cast(unsigned short, (_Float16)a);
  unsigned short ub = __builtin_bit_cast(unsigned short, (_Float16)b);
  return (unsigned int)ua | ((unsigned int)ub << 16);
}

// wave32 xor-shuffle via ds_swizzle (group-of-32 mode: and=0x1f, xor=XM)
template <int XM>
static __device__ __forceinline__ float swz_xor(float v) {
  int x = __builtin_amdgcn_ds_swizzle(__builtin_bit_cast(int, v),
                                      (XM << 10) | 0x1f);
  return __builtin_bit_cast(float, x);
}

// Flat->LDS byte offset: generic pointer to __shared__ is {aperture, offset32};
// the low 32 bits are the LDS byte address.
static __device__ __forceinline__ unsigned lds_off(const void* p) {
  return (unsigned)(unsigned long long)p;
}

// One async 16B copy global->LDS (per-lane), tracked by ASYNCcnt.
static __device__ __forceinline__ void async_cp16(unsigned lds_byte,
                                                  const _Float16* base,
                                                  unsigned gbyte) {
  asm volatile("global_load_async_to_lds_b128 %0, %1, %2"
               :: "v"(lds_byte), "v"(gbyte), "s"(base) : "memory");
}

// ---------------------------------------------------------------------------
// Fused flash-attention.  Block = 4 waves; each wave owns a 16-query tile and
// the full 384-wide output (24 f32 WMMA accumulator tiles, >256 VGPRs via the
// gfx1250 VGPR-MSB mechanism).  Q fragments are hoisted into registers for the
// whole key loop.  K/V^T 32-key blocks are double-buffered in LDS via
// GLOBAL_LOAD_ASYNC_TO_LDS_B128 so the copy of block k+1 overlaps the compute
// of block k; retirement via s_wait_asynccnt.
// ---------------------------------------------------------------------------
__global__ __launch_bounds__(128) void attn_kernel(
    const _Float16* __restrict__ qg,   // [B][N][C]
    const _Float16* __restrict__ kg,   // [B][N][C]
    const _Float16* __restrict__ vtg,  // [B][C][N]
    float* __restrict__ out)           // [B][C][N]
{
  __shared__ __align__(16) _Float16 kbuf[2][32 * 392];    // K block  [m][c]
  __shared__ __align__(16) _Float16 vbuf[2][384 * 40];    // V^T block [d][m]
  __shared__ __align__(16) _Float16 pbuf[4 * 16 * 40];    // per-wave P transpose
  const int tid  = threadIdx.x;
  const int wave = tid >> 5;
  const int lane = tid & 31;
  const int l16  = lane & 15;
  const int hb   = lane >> 4;
  const int b    = blockIdx.y;
  const int n0   = (blockIdx.x * 4 + wave) * 16;
  const _Float16* qrow = qg + ((size_t)b * NPIX + n0) * DIMC;
  const _Float16* kb   = kg + (size_t)b * NPIX * DIMC;
  const _Float16* vb   = vtg + (size_t)b * DIMC * NPIX;
  _Float16* pw = pbuf + wave * 16 * 40;

  const unsigned kofs[2] = {lds_off(&kbuf[0][0]), lds_off(&kbuf[1][0])};
  const unsigned vofs[2] = {lds_off(&vbuf[0][0]), lds_off(&vbuf[1][0])};

  // Q strip: 16 rows x 384 -> 12 A-fragments held in registers across the
  // entire key loop (96 VGPRs; the register file is 1024 VGPRs on gfx1250).
  v16h qf[12];
#pragma unroll
  for (int ck = 0; ck < 12; ++ck)
    qf[ck] = load_frag(qrow + l16 * DIMC + ck * 32 + hb * 8);

  v8f o[24];
  v8f zero = {0.f, 0.f, 0.f, 0.f, 0.f, 0.f, 0.f, 0.f};
#pragma unroll
  for (int i = 0; i < 24; ++i) o[i] = zero;
  float mi[8], li[8];
#pragma unroll
  for (int r = 0; r < 8; ++r) { mi[r] = -3.0e38f; li[r] = 0.f; }

  // Issue async staging of key-block `m0` into buffer `bsel`.
  auto stage = [&](int bsel, int m0) {
#pragma unroll
    for (int it = 0; it < 12; ++it) {            // K: 32 x 384 halves
      int chunk = it * 128 + tid;
      int m = chunk / 48, off = (chunk % 48) * 8;
      async_cp16(kofs[bsel] + (unsigned)(m * 392 + off) * 2, kb,
                 (unsigned)(((m0 + m) * DIMC + off) * 2));
    }
#pragma unroll
    for (int it = 0; it < 12; ++it) {            // V^T: 384 x 32 halves
      int chunk = it * 128 + tid;
      int d = chunk >> 2, off = (chunk & 3) * 8;
      async_cp16(vofs[bsel] + (unsigned)(d * 40 + off) * 2, vb,
                 (unsigned)((d * NPIX + m0 + off) * 2));
    }
  };

  stage(0, 0);                     // prefetch first key block
  for (int kbk = 0; kbk < 98; ++kbk) {
    const int cur = kbk & 1;
    __syncthreads();               // all waves done reading buf[cur^1]
    if (kbk < 97) {
      stage(cur ^ 1, (kbk + 1) * 32);
      // own 24 older asyncs (current block) retired; 24 new may be in flight
      asm volatile("s_wait_asynccnt 0x18" ::: "memory");
    } else {
      asm volatile("s_wait_asynccnt 0x0" ::: "memory");
    }
    __syncthreads();               // current block visible to all waves

    const _Float16* kcur = &kbuf[cur][0];
    const _Float16* vcur = &vbuf[cur][0];

    // S = Q @ K^T : two 16x16 f32 tiles over this 32-key block.
    v8f s0 = zero, s1 = zero;
#pragma unroll
    for (int ck = 0; ck < 12; ++ck) {
      v16h k0 = load_frag(kcur + l16 * 392 + ck * 32 + hb * 8);
      v16h k1 = load_frag(kcur + (16 + l16) * 392 + ck * 32 + hb * 8);
      s0 = wmma16(qf[ck], k0, s0);
      s1 = wmma16(qf[ck], k1, s1);
    }

    // Online softmax (rows r+8*hb; reduce across the 16 lanes of each half).
    float sc[8];
#pragma unroll
    for (int r = 0; r < 8; ++r) {
      float mv = fmaxf(s0[r], s1[r]);
      mv = fmaxf(mv, swz_xor<1>(mv));
      mv = fmaxf(mv, swz_xor<2>(mv));
      mv = fmaxf(mv, swz_xor<4>(mv));
      mv = fmaxf(mv, swz_xor<8>(mv));
      float mn = fmaxf(mi[r], mv);
      float s  = __expf(mi[r] - mn);
      mi[r] = mn;
      float p0 = __expf(s0[r] - mn);
      float p1 = __expf(s1[r] - mn);
      float rs = p0 + p1;
      rs += swz_xor<1>(rs);
      rs += swz_xor<2>(rs);
      rs += swz_xor<4>(rs);
      rs += swz_xor<8>(rs);
      li[r] = li[r] * s + rs;
      sc[r] = s;
      s0[r] = p0; s1[r] = p1;
    }
#pragma unroll
    for (int dt = 0; dt < 24; ++dt) {
#pragma unroll
      for (int r = 0; r < 8; ++r) o[dt][r] *= sc[r];
    }

    // P -> f16, LDS transpose into A-layout (row-major [n][m], stride 40).
#pragma unroll
    for (int r = 0; r < 8; ++r) {
      pw[(r + 8 * hb) * 40 + l16]      = (_Float16)s0[r];
      pw[(r + 8 * hb) * 40 + 16 + l16] = (_Float16)s1[r];
    }
    v16h pf = load_frag(pw + l16 * 40 + hb * 8);

    // O += P @ V  (24 d-tiles, K = 32 keys)
#pragma unroll
    for (int dt = 0; dt < 24; ++dt) {
      v16h vf = load_frag(vcur + (dt * 16 + l16) * 40 + hb * 8);
      o[dt] = wmma16(pf, vf, o[dt]);
    }
  }

  // Epilogue: divide by softmax sum, store out[b][d][n] (contiguous n/lane).
  float inv[8];
#pragma unroll
  for (int r = 0; r < 8; ++r) inv[r] = 1.f / li[r];
  float* ob = out + (size_t)b * DIMC * NPIX;
#pragma unroll
  for (int dt = 0; dt < 24; ++dt) {
    int d = dt * 16 + l16;
    float* orow = ob + (size_t)d * NPIX + n0 + hb * 8;
    float4 f0, f1;
    f0.x = o[dt][0] * inv[0]; f0.y = o[dt][1] * inv[1];
    f0.z = o[dt][2] * inv[2]; f0.w = o[dt][3] * inv[3];
    f1.x = o[dt][4] * inv[4]; f1.y = o[dt][5] * inv[5];
    f1.z = o[dt][6] * inv[6]; f1.w = o[dt][7] * inv[7];
    *(float4*)(orow)     = f0;
    *(float4*)(orow + 4) = f1;
  }
}

// ---------------------------------------------------------------------------
// Kernel 1: q = (x^T @ W1^T) * C^-0.5   -> f16 [B][N][C]
//           k = f16(x^T)                -> f16 [B][N][C]
//           v^T = (x^T @ W2^T)^T        -> f16 [B][C][N]
// One wave per 16-token strip; block = 2 waves.
// ---------------------------------------------------------------------------
__global__ __launch_bounds__(64) void proj_kernel(
    const float* __restrict__ x,    // [B][C][N]
    const float* __restrict__ w1,   // [C][C]
    const float* __restrict__ w2,   // [C][C]
    _Float16* __restrict__ qg,      // [B][N][C]
    _Float16* __restrict__ kg,      // [B][N][C]
    _Float16* __restrict__ vtg)     // [B][C][N]
{
  __shared__ __align__(16) _Float16 tl[2 * 16 * 392];  // t strip, [n][c] pad 392
  __shared__ __align__(16) _Float16 qs[2 * 16 * 16];   // per-wave q transpose bounce
  const int wave = threadIdx.x >> 5;
  const int lane = threadIdx.x & 31;
  const int l16  = lane & 15;
  const int hb   = lane >> 4;
  const int b    = blockIdx.y;
  const int n0   = (blockIdx.x * 2 + wave) * 16;
  _Float16* tw = tl + wave * 16 * 392;
  _Float16* qw = qs + wave * 256;
  const float* xb = x + (size_t)b * DIMC * NPIX;

  // Stage 16 x 384 t-strip into LDS as f16 (coalesced b128 along n).
  {
    const int cl = lane >> 2;        // 0..7 : c sub-row
    const int ch = (lane & 3) * 4;   // 0,4,8,12 : n sub-offset
    for (int it = 0; it < 48; ++it) {
      int c = it * 8 + cl;
      const float4 f = *(const float4*)(xb + (size_t)c * NPIX + n0 + ch);
      tw[(ch + 0) * 392 + c] = (_Float16)f.x;
      tw[(ch + 1) * 392 + c] = (_Float16)f.y;
      tw[(ch + 2) * 392 + c] = (_Float16)f.z;
      tw[(ch + 3) * 392 + c] = (_Float16)f.w;
    }
  }
  // Emit K rows (coalesced b128).
  {
    _Float16* kout = kg + ((size_t)b * NPIX + n0) * DIMC;
    for (int it = 0; it < 24; ++it) {
      int chunk = it * 32 + lane;               // 768 chunks x 8 halves
      int r = chunk / 48, off = (chunk % 48) * 8;
      v4u t = *(const v4u*)(tw + r * 392 + off);
      *(v4u*)(kout + r * DIMC + off) = t;
    }
  }

  const float kScale = 0.051031036307982884f;   // 384^-0.5 folded into q
  for (int dt = 0; dt < 24; ++dt) {
    v8f aq = {0.f, 0.f, 0.f, 0.f, 0.f, 0.f, 0.f, 0.f};
    v8f av = aq;
    const int d = dt * 16 + l16;
    const float* w1r = w1 + (size_t)d * DIMC;
    const float* w2r = w2 + (size_t)d * DIMC;
    for (int ck = 0; ck < 12; ++ck) {
      v16h af = load_frag(tw + l16 * 392 + ck * 32 + hb * 8);
      v16h b1 = wfrag(w1r + ck * 32, hb);
      v16h b2 = wfrag(w2r + ck * 32, hb);
      aq = wmma16(af, b1, aq);
      av = wmma16(af, b2, av);
    }
    // v^T store: lane holds d=dt*16+l16, 8 consecutive n -> one b128.
    {
      v4u pk;
      pk[0] = pack2(av[0], av[1]);
      pk[1] = pack2(av[2], av[3]);
      pk[2] = pack2(av[4], av[5]);
      pk[3] = pack2(av[6], av[7]);
      *(v4u*)(vtg + ((size_t)b * DIMC + d) * NPIX + n0 + hb * 8) = pk;
    }
    // q store via LDS transpose bounce (row-major [n][d]).
#pragma unroll
    for (int r = 0; r < 8; ++r)
      qw[(r + 8 * hb) * 16 + l16] = (_Float16)(aq[r] * kScale);
    {
      v4u t = *(const v4u*)(qw + (lane >> 1) * 16 + (lane & 1) * 8);
      *(v4u*)(qg + ((size_t)b * NPIX + n0 + (lane >> 1)) * DIMC + dt * 16 +
              (lane & 1) * 8) = t;
    }
  }
}

extern "C" void kernel_launch(void* const* d_in, const int* in_sizes, int n_in,
                              void* d_out, int out_size, void* d_ws, size_t ws_size,
                              hipStream_t stream) {
  (void)in_sizes; (void)n_in; (void)out_size; (void)ws_size;
  const float* x  = (const float*)d_in[0];
  const float* w1 = (const float*)d_in[1];
  const float* w2 = (const float*)d_in[2];
  float* out = (float*)d_out;

  size_t sz = (size_t)NBATCH * NPIX * DIMC;     // halves per f16 buffer
  _Float16* qg  = (_Float16*)d_ws;              // [B][N][C] f16
  _Float16* kg  = qg + sz;                      // [B][N][C] f16
  _Float16* vtg = kg + sz;                      // [B][C][N] f16

  dim3 g1(98, NBATCH), b1(64);
  proj_kernel<<<g1, b1, 0, stream>>>(x, w1, w2, qg, kg, vtg);
  dim3 g2(49, NBATCH), b2(128);
  attn_kernel<<<g2, b2, 0, stream>>>(qg, kg, vtg, out);
}
